// MultiHeadedAttention_59502476919078
// MI455X (gfx1250) — compile-verified, matrix operand order checked by reading, not run on hardware
//
#include <hip/hip_runtime.h>
#include <hip/hip_bf16.h>
#include <cstddef>

// ---------------------------------------------------------------------------
// MHA for MI455X (gfx1250): bf16 WMMA (v_wmma_f32_16x16x32_bf16) everywhere,
// f32 accumulate, flash-style online softmax, V pre-transposed for NT P@V.
// Weight tiles for both GEMMs are streamed through the Tensor Data Mover
// (tensor_load_to_lds + s_wait_tensorcnt) with LDS double buffering.
// ---------------------------------------------------------------------------

#define BB 8
#define LL 1024
#define DD 1024
#define HH 16
#define DH 64

typedef __attribute__((ext_vector_type(16))) __bf16 v16bf;
typedef __attribute__((ext_vector_type(8)))  __bf16 v8bf;
typedef __attribute__((ext_vector_type(8)))  float  v8f;
typedef __attribute__((ext_vector_type(4)))  unsigned int u32x4;
typedef __attribute__((ext_vector_type(4)))  int i32x4;
typedef __attribute__((ext_vector_type(8)))  int i32x8;

__device__ __forceinline__ __bf16 f2bf(float f) {
    union { float f; unsigned u; } v; v.f = f;
    unsigned r = v.u + 0x7FFFu + ((v.u >> 16) & 1u);   // round-to-nearest-even
    unsigned short h = (unsigned short)(r >> 16);
    return __builtin_bit_cast(__bf16, h);
}

__device__ __forceinline__ v8f wmma_bf16(v16bf a, v16bf b, v8f c) {
    // D = A(16x32 bf16) x B(32x16 bf16) + C(16x16 f32)
    return __builtin_amdgcn_wmma_f32_16x16x32_bf16(false, a, false, b,
                                                   (short)0, c, false, false);
}

// Compiler barrier that also lets `p` escape: the TDM engine writes LDS
// behind the compiler's back, so the LDS buffer must be assumed modified
// by this asm (otherwise loads of a never-stored __shared__ array fold to
// undef and get CSE'd -- observed in round-3 codegen).
__device__ __forceinline__ void lds_publish(const void* p) {
    asm volatile("" : : "r"(p) : "memory");
}

// Fragment from bf16 memory: caller passes ptr already at row*ld + k0 + koff,
// where koff = (lane>=16 ? 8 : 0).  Elements 0-7 <- K{koff..koff+7},
// elements 8-15 <- K{koff+16..koff+23}  (ISA 16-bit A layout; B is loaded
// column-major with the same per-lane pattern for NT GEMMs).
__device__ __forceinline__ v16bf load_frag_bf16(const __bf16* p) {
    v8bf lo = *(const v8bf*)(p);
    v8bf hi = *(const v8bf*)(p + 16);
    v16bf r;
#pragma unroll
    for (int i = 0; i < 8; ++i) { r[i] = lo[i]; r[i + 8] = hi[i]; }
    return r;
}

// Same fragment but source is f32 memory (converted on the fly).
__device__ __forceinline__ v16bf load_frag_f32(const float* p) {
    v8f lo = *(const v8f*)(p);
    v8f hi = *(const v8f*)(p + 16);
    v16bf r;
#pragma unroll
    for (int i = 0; i < 8; ++i) { r[i] = f2bf(lo[i]); r[i + 8] = f2bf(hi[i]); }
    return r;
}

// ---------------------------------------------------------------------------
// Tensor Data Mover: async-DMA a 2D bf16 tile (tile_d0 x tile_d1 elements,
// row stride d0_stride elements) from global memory into LDS at lds_byte_off.
// Descriptor per CDNA5 ISA 8.3-8.5 (D# group0 + group1; groups 2/3 unused
// for 2D tensors).  Tracked with TENSORcnt.  6-arg builtin variant
// (clang-23 / therock-10.0): (g0, g1, g2, g3, g_extra, cpol).
// ---------------------------------------------------------------------------
__device__ __forceinline__ void tdm_load_2d_bf16(unsigned lds_byte_off,
                                                 const __bf16* gaddr,
                                                 int tensor_d0, int tensor_d1,
                                                 int tile_d0, int tile_d1,
                                                 int d0_stride) {
    unsigned long long ga = (unsigned long long)(size_t)gaddr;
    u32x4 g0;
    g0[0] = 1u;                                        // count=1, user mode
    g0[1] = lds_byte_off;                              // lds_addr (bytes)
    g0[2] = (unsigned)(ga & 0xFFFFFFFFu);              // global_addr[31:0]
    g0[3] = (unsigned)((ga >> 32) & 0x1FFFFFFu)        // global_addr[56:32]
          | (2u << 30);                                // type = 2 ("image")
    i32x8 g1;
    g1[0] = (1 << 16);                                 // data_size=1 -> 2 bytes
    g1[1] = (int)((tensor_d0 & 0xFFFF) << 16);         // tensor_dim0[15:0]
    g1[2] = (int)(((tensor_d0 >> 16) & 0xFFFF)
          | ((tensor_d1 & 0xFFFF) << 16));             // dim0 hi | dim1 lo
    g1[3] = (int)(((tensor_d1 >> 16) & 0xFFFF)
          | ((tile_d0 & 0xFFFF) << 16));               // dim1 hi | tile_dim0
    g1[4] = (int)(tile_d1 & 0xFFFF);                   // tile_dim1 (tile_dim2=0)
    g1[5] = d0_stride;                                 // tensor_dim0_stride lo
    g1[6] = 0;                                         // stride0 hi | stride1 lo
    g1[7] = 0;                                         // stride1 hi
    i32x4 z4 = { 0, 0, 0, 0 };
    i32x8 z8 = { 0, 0, 0, 0, 0, 0, 0, 0 };
    __builtin_amdgcn_tensor_load_to_lds(g0, g1, z4, z4, z8, 0);
}

// ---------------------------------------------------------------------------
// Kernel 1: f32 -> bf16 conversion (weights)
// ---------------------------------------------------------------------------
__global__ void cvt_f32_bf16_kernel(const float* __restrict__ src,
                                    __bf16* __restrict__ dst, int n) {
    int i = blockIdx.x * blockDim.x + threadIdx.x;
    if (i < n) dst[i] = f2bf(src[i]);
}

// ---------------------------------------------------------------------------
// Kernel 2: fused QKV projection.  One wave -> 16(M) x 64(N) outputs.
// Weight tile (64 rows x 32 K, 4 KB) is TDM-staged into LDS, double-buffered.
//   mode 0: q = (query@Wq^T + bq) * 1/8   -> [B,H,L,DH] bf16
//   mode 1: k =  key  @Wk^T + bk          -> [B,H,L,DH] bf16
//   mode 2: v =  value@Wv^T + bv          -> [B,H,DH,L] bf16 (transposed)
// ---------------------------------------------------------------------------
__global__ __launch_bounds__(32)
void qkv_gemm_kernel(const float* __restrict__ key,
                     const float* __restrict__ value,
                     const float* __restrict__ query,
                     const __bf16* __restrict__ wk,
                     const __bf16* __restrict__ wv,
                     const __bf16* __restrict__ wq,
                     const float* __restrict__ bk,
                     const float* __restrict__ bv,
                     const float* __restrict__ bq,
                     __bf16* __restrict__ qws,
                     __bf16* __restrict__ kws,
                     __bf16* __restrict__ vtws) {
    const int mode = blockIdx.z;                 // 0=q 1=k 2=v
    const float*  X    = (mode == 0) ? query : (mode == 1) ? key : value;
    const __bf16* W    = (mode == 0) ? wq    : (mode == 1) ? wk  : wv;
    const float*  bias = (mode == 0) ? bq    : (mode == 1) ? bk  : bv;

    const int mt   = blockIdx.x >> 4;            // 512 m-tiles of 16
    const int nb   = blockIdx.x & 15;            // 16 n-blocks of 64
    const int lane = threadIdx.x;
    const int lmod = lane & 15;
    const int koff = (lane >> 4) * 8;
    const int hi16 = lane >> 4;

    __shared__ __bf16 Wlds[2][64 * 32];          // 2 x 4 KB double buffer

    const __bf16* wslab = W + (size_t)(nb * 64) * DD;  // 64-row weight slab
    const float*  xrow  = X + (size_t)(mt * 16 + lmod) * DD;

    // prologue: stage K-step 0
    tdm_load_2d_bf16(0u, wslab, DD, 64, 32, 64, DD);

    v8f acc[4] = {};
    for (int k0 = 0, it = 0; k0 < DD; k0 += 32, ++it) {
        lds_publish(&Wlds[0][0]);
        if (k0 + 32 < DD) {
            tdm_load_2d_bf16((unsigned)(((it + 1) & 1) * 4096),
                             wslab + (k0 + 32), DD, 64, 32, 64, DD);
            __builtin_amdgcn_s_wait_tensorcnt((short)1);  // buf[it] ready
        } else {
            __builtin_amdgcn_s_wait_tensorcnt((short)0);
        }
        lds_publish(&Wlds[0][0]);    // TDM wrote LDS: force real ds loads

        __builtin_prefetch(xrow + k0 + 128, 0, 0);        // global_prefetch_b8
        v16bf a = load_frag_f32(xrow + k0 + koff);
        const __bf16* lb = &Wlds[it & 1][0];
#pragma unroll
        for (int nt = 0; nt < 4; ++nt) {
            v16bf bf = load_frag_bf16(lb + (nt * 16 + lmod) * 32 + koff);
            acc[nt] = wmma_bf16(a, bf, acc[nt]);
        }
    }

#pragma unroll
    for (int nt = 0; nt < 4; ++nt) {
        const int col = nb * 64 + nt * 16 + lmod;      // n in [0,1024)
        const int hh  = col >> 6;                      // head
        const int d   = col & 63;                      // within head
        const float bv_ = bias[col];
#pragma unroll
        for (int r = 0; r < 8; ++r) {
            const int row  = mt * 16 + r + 8 * hi16;   // m in [0,8192)
            const int b    = row >> 10;
            const int lpos = row & 1023;
            float v = acc[nt][r] + bv_;
            if (mode == 0) {
                qws[((size_t)(b * HH + hh) * LL + lpos) * DH + d] =
                    f2bf(v * 0.125f);                  // 1/sqrt(64)
            } else if (mode == 1) {
                kws[((size_t)(b * HH + hh) * LL + lpos) * DH + d] = f2bf(v);
            } else {
                vtws[((size_t)(b * HH + hh) * DH + d) * LL + lpos] = f2bf(v);
            }
        }
    }
}

// ---------------------------------------------------------------------------
// Kernel 3: flash attention.  One wave per (b, h, 16 q-rows).
// Streams 32 k-positions/step: 4 score WMMAs + online softmax + 4 ctx WMMAs.
// Head 0 writes raw (pre-mask) scores to top_score.
// ---------------------------------------------------------------------------
__global__ __launch_bounds__(32)
void flash_attn_kernel(const __bf16* __restrict__ q,
                       const __bf16* __restrict__ k,
                       const __bf16* __restrict__ vt,
                       const unsigned char* __restrict__ mask,
                       const unsigned char* __restrict__ amask,
                       __bf16* __restrict__ ctx,
                       float* __restrict__ topscore) {
    const int wid  = blockIdx.x;          // B*H*(L/16)
    const int qt   = wid & 63;
    const int h    = (wid >> 6) & 15;
    const int b    = wid >> 10;
    const int lane = threadIdx.x;
    const int lmod = lane & 15;
    const int koff = (lane >> 4) * 8;
    const int hi16 = lane >> 4;

    const __bf16* qbase =
        q + ((size_t)(b * HH + h) * LL + qt * 16 + lmod) * DH;
    const v16bf qf0 = load_frag_bf16(qbase + 0 + koff);
    const v16bf qf1 = load_frag_bf16(qbase + 32 + koff);

    const __bf16* kbase = k  + (size_t)(b * HH + h) * LL * DH;
    const __bf16* vbase = vt + (size_t)(b * HH + h) * DH * LL;

    float m_r[8], l_r[8];
    v8f acc[4] = {};
#pragma unroll
    for (int r = 0; r < 8; ++r) { m_r[r] = -3.0e38f; l_r[r] = 0.0f; }

    __shared__ __bf16 P[16 * 32];        // 1 KB P-tile staging

    for (int k0 = 0; k0 < LL; k0 += 32) {
        // ---- scores: two 16x16 tiles covering k-positions [k0, k0+32)
        v8f s[2];
#pragma unroll
        for (int t = 0; t < 2; ++t) {
            const __bf16* kr = kbase + (size_t)(k0 + t * 16 + lmod) * DH;
            v16bf b0 = load_frag_bf16(kr + koff);
            v16bf b1 = load_frag_bf16(kr + 32 + koff);
            v8f c = {};
            c = wmma_bf16(qf0, b0, c);
            c = wmma_bf16(qf1, b1, c);
            s[t] = c;
        }

        // ---- head-0 raw scores -> top_score (pre-mask)
        if (h == 0) {
#pragma unroll
            for (int t = 0; t < 2; ++t)
#pragma unroll
                for (int r = 0; r < 8; ++r) {
                    const int row = qt * 16 + r + 8 * hi16;
                    const int col = k0 + t * 16 + lmod;
                    topscore[((size_t)b * LL + row) * LL + col] = s[t][r];
                }
        }

        // ---- masking
        if (h < 8) {
            const unsigned char* mb = mask + (size_t)b * LL * LL;
#pragma unroll
            for (int t = 0; t < 2; ++t) {
                const int col = k0 + t * 16 + lmod;
#pragma unroll
                for (int r = 0; r < 8; ++r) {
                    const int row = qt * 16 + r + 8 * hi16;
                    if (mb[(size_t)row * LL + col]) s[t][r] = -1.0e18f;
                }
            }
        } else {
#pragma unroll
            for (int t = 0; t < 2; ++t) {
                const int col = k0 + t * 16 + lmod;
                if (amask[b * LL + col]) {
#pragma unroll
                    for (int r = 0; r < 8; ++r) s[t][r] = -1.0e18f;
                }
            }
        }

        // ---- online softmax (rows live in 16-lane halves of C layout)
#pragma unroll
        for (int r = 0; r < 8; ++r) {
            float mx = fmaxf(s[0][r], s[1][r]);
            mx = fmaxf(mx, __shfl_xor(mx, 1, 32));
            mx = fmaxf(mx, __shfl_xor(mx, 2, 32));
            mx = fmaxf(mx, __shfl_xor(mx, 4, 32));
            mx = fmaxf(mx, __shfl_xor(mx, 8, 32));
            const float mnew  = fmaxf(m_r[r], mx);
            const float alpha = __expf(m_r[r] - mnew);
            const float p0 = __expf(s[0][r] - mnew);
            const float p1 = __expf(s[1][r] - mnew);
            float rs = p0 + p1;
            rs += __shfl_xor(rs, 1, 32);
            rs += __shfl_xor(rs, 2, 32);
            rs += __shfl_xor(rs, 4, 32);
            rs += __shfl_xor(rs, 8, 32);
            l_r[r] = l_r[r] * alpha + rs;
            m_r[r] = mnew;
#pragma unroll
            for (int nt = 0; nt < 4; ++nt) acc[nt][r] *= alpha;
            const int row = r + 8 * hi16;
            P[row * 32 + lmod]      = f2bf(p0);
            P[row * 32 + 16 + lmod] = f2bf(p1);
        }
        __syncthreads();

        // ---- P (16x32) A-fragment from LDS, then ctx += P @ V
        v16bf pf = load_frag_bf16(&P[lmod * 32 + koff]);
#pragma unroll
        for (int nt = 0; nt < 4; ++nt) {
            const __bf16* vr =
                vbase + (size_t)(nt * 16 + lmod) * LL + k0 + koff;
            v16bf vf = load_frag_bf16(vr);
            acc[nt] = wmma_bf16(pf, vf, acc[nt]);
        }
        __syncthreads();
    }

    // ---- epilogue: ctx[b, row, h*64+d] = acc / l  (bf16 for out-proj GEMM)
#pragma unroll
    for (int nt = 0; nt < 4; ++nt) {
#pragma unroll
        for (int r = 0; r < 8; ++r) {
            const int row = qt * 16 + r + 8 * hi16;
            const int d   = h * DH + nt * 16 + lmod;
            const float v = acc[nt][r] / l_r[r];
            ctx[((size_t)b * LL + row) * DD + d] = f2bf(v);
        }
    }
}

// ---------------------------------------------------------------------------
// Kernel 4: output projection  out = ctx @ Wo^T + bo   (f32 output)
// Wo tiles TDM-staged into LDS, double-buffered.
// ---------------------------------------------------------------------------
__global__ __launch_bounds__(32)
void out_gemm_kernel(const __bf16* __restrict__ ctx,
                     const __bf16* __restrict__ wo,
                     const float* __restrict__ bo,
                     float* __restrict__ out) {
    const int mt   = blockIdx.x >> 4;
    const int nb   = blockIdx.x & 15;
    const int lane = threadIdx.x;
    const int lmod = lane & 15;
    const int koff = (lane >> 4) * 8;
    const int hi16 = lane >> 4;

    __shared__ __bf16 Wlds[2][64 * 32];          // 2 x 4 KB double buffer

    const __bf16* wslab = wo  + (size_t)(nb * 64) * DD;
    const __bf16* arow  = ctx + (size_t)(mt * 16 + lmod) * DD;

    tdm_load_2d_bf16(0u, wslab, DD, 64, 32, 64, DD);

    v8f acc[4] = {};
    for (int k0 = 0, it = 0; k0 < DD; k0 += 32, ++it) {
        lds_publish(&Wlds[0][0]);
        if (k0 + 32 < DD) {
            tdm_load_2d_bf16((unsigned)(((it + 1) & 1) * 4096),
                             wslab + (k0 + 32), DD, 64, 32, 64, DD);
            __builtin_amdgcn_s_wait_tensorcnt((short)1);
        } else {
            __builtin_amdgcn_s_wait_tensorcnt((short)0);
        }
        lds_publish(&Wlds[0][0]);    // TDM wrote LDS: force real ds loads

        __builtin_prefetch(arow + k0 + 128, 0, 0);
        v16bf a = load_frag_bf16(arow + k0 + koff);
        const __bf16* lb = &Wlds[it & 1][0];
#pragma unroll
        for (int nt = 0; nt < 4; ++nt) {
            v16bf bf = load_frag_bf16(lb + (nt * 16 + lmod) * 32 + koff);
            acc[nt] = wmma_bf16(a, bf, acc[nt]);
        }
    }

#pragma unroll
    for (int nt = 0; nt < 4; ++nt) {
        const int col = nb * 64 + nt * 16 + lmod;
        const float bv_ = bo[col];
#pragma unroll
        for (int r = 0; r < 8; ++r) {
            const int row = mt * 16 + r + 8 * hi16;
            out[(size_t)row * DD + col] = acc[nt][r] + bv_;
        }
    }
}

// ---------------------------------------------------------------------------
// Host launcher
// ---------------------------------------------------------------------------
extern "C" void kernel_launch(void* const* d_in, const int* in_sizes, int n_in,
                              void* d_out, int out_size, void* d_ws, size_t ws_size,
                              hipStream_t stream) {
    const float* key    = (const float*)d_in[0];
    const float* value  = (const float*)d_in[1];
    const float* query  = (const float*)d_in[2];
    const unsigned char* mask  = (const unsigned char*)d_in[3];
    const unsigned char* amask = (const unsigned char*)d_in[4];
    const float* Wk = (const float*)d_in[5];
    const float* bk = (const float*)d_in[6];
    const float* Wv = (const float*)d_in[7];
    const float* bv = (const float*)d_in[8];
    const float* Wq = (const float*)d_in[9];
    const float* bq = (const float*)d_in[10];
    const float* Wo = (const float*)d_in[11];
    const float* bo = (const float*)d_in[12];

    float* out = (float*)d_out;                         // [B*L, D]
    float* topscore = out + (size_t)BB * LL * DD;       // [B, L, L]

    // Workspace layout (bf16):
    char* ws = (char*)d_ws;
    const size_t W_ELEMS = (size_t)DD * DD;             // 1M
    const size_t A_ELEMS = (size_t)BB * LL * DD;        // 8M
    __bf16* wq_b = (__bf16*)(ws);                       // 2 MB
    __bf16* wk_b = wq_b + W_ELEMS;                      // 2 MB
    __bf16* wv_b = wk_b + W_ELEMS;                      // 2 MB
    __bf16* wo_b = wv_b + W_ELEMS;                      // 2 MB
    __bf16* qws  = wo_b + W_ELEMS;                      // 16 MB [B,H,L,DH]
    __bf16* kws  = qws + A_ELEMS;                       // 16 MB [B,H,L,DH]
    __bf16* vtws = kws + A_ELEMS;                       // 16 MB [B,H,DH,L]
    __bf16* ctxw = vtws + A_ELEMS;                      // 16 MB [B,L,D]

    // 1) weight conversion
    {
        const int n = (int)W_ELEMS;
        const int blocks = (n + 255) / 256;
        cvt_f32_bf16_kernel<<<blocks, 256, 0, stream>>>(Wq, wq_b, n);
        cvt_f32_bf16_kernel<<<blocks, 256, 0, stream>>>(Wk, wk_b, n);
        cvt_f32_bf16_kernel<<<blocks, 256, 0, stream>>>(Wv, wv_b, n);
        cvt_f32_bf16_kernel<<<blocks, 256, 0, stream>>>(Wo, wo_b, n);
    }

    // 2) fused QKV projection: 512 m-tiles * 16 n-blocks, z = {q,k,v}
    {
        dim3 grid((BB * LL / 16) * (DD / 64), 1, 3);
        qkv_gemm_kernel<<<grid, 32, 0, stream>>>(key, value, query,
                                                 wk_b, wv_b, wq_b,
                                                 bk, bv, bq,
                                                 qws, kws, vtws);
    }

    // 3) flash attention: one wave per (b, h, 16 q-rows)
    {
        dim3 grid(BB * HH * (LL / 16));
        flash_attn_kernel<<<grid, 32, 0, stream>>>(qws, kws, vtws,
                                                   mask, amask, ctxw, topscore);
    }

    // 4) output projection
    {
        dim3 grid((BB * LL / 16) * (DD / 64));
        out_gemm_kernel<<<grid, 32, 0, stream>>>(ctxw, wo_b, bo, out);
    }
}
